// RegionProposalNetwork_24781961298209
// MI455X (gfx1250) — matrix operand order repaired and checked by reading, not compiled, not run on hardware
//
#include <hip/hip_runtime.h>
#include <hip/hip_bf16.h>

typedef __attribute__((ext_vector_type(16))) _Float16 v16h;
typedef __attribute__((ext_vector_type(8)))  _Float16 v8h;
typedef __attribute__((ext_vector_type(8)))  float    v8f;

#define N_BATCH   16
#define C_IN      512
#define HW        64
#define PIXELS    (N_BATCH * HW * HW)        // 65536
#define ANCH_PER  (HW * HW * 9)              // 36864 per batch
#define N_ANCH_T  (PIXELS * 9)               // 589824 total
#define POST_NMS  300

// d_out float regions
#define LOC_OFF    0                          // 16*36864*4 = 2359296 floats
#define SCORE_OFF  2359296                    // 16*36864*2 = 1179648 floats
#define ROIS_OFF   3538944                    // 16*300*4   = 19200 floats

// workspace byte offsets (total ~86 MB)
#define WH_OFF     ((size_t)0)                          // 9*512*512 f16
#define WH2_OFF    (WH_OFF  + (size_t)9*512*512*2)      // 512*64 f16
#define FEAT_OFF   (WH2_OFF + (size_t)512*64*2)         // 65536*512 f16
#define FG_OFF     (FEAT_OFF + (size_t)PIXELS*512*2)    // 589824 f32
#define BOX_OFF    (FG_OFF  + (size_t)N_ANCH_T*4)       // 589824*4 f32
#define SCF_OFF    (BOX_OFF + (size_t)N_ANCH_T*16)      // 589824 f32

__device__ __forceinline__ v16h cat16(v8h a, v8h b) {
  return __builtin_shufflevector(a, b, 0,1,2,3,4,5,6,7,8,9,10,11,12,13,14,15);
}

// A fragment (16x32 f16): lane half h: e0..7 -> K = 8h+e ; e8..15 -> K = 16+8h+(e-8)
__device__ __forceinline__ v16h load_fragA(const _Float16* base, int row, int lane) {
  int half = lane >> 4;
  const _Float16* p = base + row * 32;
  v8h lo = *(const v8h*)(p + 8 * half);
  v8h hi = *(const v8h*)(p + 16 + 8 * half);
  return cat16(lo, hi);
}
// B fragment (32x16 f16, stored transposed [N][K] in LDS): lane half h: e0..7 -> K = 8h+e ; e8..15 -> K = 16+8h+(e-8)
__device__ __forceinline__ v16h load_fragB(const _Float16* base, int col, int lane) {
  int half = lane >> 4;
  const _Float16* p = base + col * 32;
  v8h lo = *(const v8h*)(p + 8 * half);
  v8h hi = *(const v8h*)(p + 16 + 8 * half);
  return cat16(lo, hi);
}

// ---------------- weight repack ----------------
__global__ void prep_w_kernel(const float* __restrict__ w, _Float16* __restrict__ wh) {
  int t = blockIdx.x * blockDim.x + threadIdx.x;
  if (t >= 9 * 512 * 512) return;
  int tap = t / (512 * 512);
  int r   = t - tap * (512 * 512);
  int ci  = r >> 9, co = r & 511;
  wh[t] = (_Float16)w[((size_t)co * 512 + ci) * 9 + tap];
}

__global__ void prep_w2_kernel(const float* __restrict__ loc_w, const float* __restrict__ score_w,
                               _Float16* __restrict__ wh2) {
  int t = blockIdx.x * blockDim.x + threadIdx.x;
  if (t >= 512 * 64) return;
  int ci = t >> 6, co = t & 63;
  float v = 0.f;
  if (co < 36)      v = loc_w[(size_t)co * 512 + ci];
  else if (co < 54) v = score_w[(size_t)(co - 36) * 512 + ci];
  wh2[(size_t)ci * 64 + co] = (_Float16)v;
}

// ---------------- 3x3 conv + ReLU, implicit GEMM on WMMA ----------------
// grid (1024, 4), block 256 (8 wave32). M tile = one (n,h) row (64 px), N tile = 128 co.
// Double-buffered LDS pipeline: one barrier per K-step.

__device__ __forceinline__ void conv_stageA(_Float16* A, const float* __restrict__ x,
                                            int n, int h, int it, int tid) {
  const int tap = it >> 4;             // 16 K-chunks per tap, 9 taps
  const int ck  = it & 15;
  const int kh = tap / 3, kw = tap - 3 * kh;
  const int yin = h + kh - 1;
  const bool yok = (yin >= 0) && (yin < 64);
  const int c = tid >> 3, wseg = tid & 7;        // 32 channels x 8 w-segments
  const float* srcc = x + (size_t)n * (512 * 4096) + (size_t)(ck * 32 + c) * 4096;
  if (ck < 15)
    __builtin_prefetch(srcc + (size_t)32 * 4096 + (yok ? yin : 0) * 64, 0, 1);
  #pragma unroll
  for (int i2 = 0; i2 < 8; ++i2) {
    const int w = wseg * 8 + i2;
    const int win = w + kw - 1;
    float v = 0.f;
    if (yok && (unsigned)win < 64u) v = srcc[yin * 64 + win];
    A[w * 32 + c] = (_Float16)v;
  }
}

__device__ __forceinline__ void conv_stageB(_Float16* B, const _Float16* __restrict__ wh,
                                            int nq, int it, int tid) {
  const int tap = it >> 4;
  const int ck  = it & 15;
  const int c = tid >> 3, jseg = tid & 7;        // 32 channels x 8 co-segments of 16
  const _Float16* src = wh + (size_t)tap * (512 * 512) + (size_t)(ck * 32 + c) * 512
                           + nq * 128 + jseg * 16;
  v8h p0 = *(const v8h*)src;
  v8h p1 = *(const v8h*)(src + 8);
  #pragma unroll
  for (int i = 0; i < 8; ++i) B[(jseg * 16 + i) * 32 + c] = p0[i];
  #pragma unroll
  for (int i = 0; i < 8; ++i) B[(jseg * 16 + 8 + i) * 32 + c] = p1[i];
}

__global__ __launch_bounds__(256) void conv3x3_wmma_kernel(
    const float* __restrict__ x, const float* __restrict__ bias,
    const _Float16* __restrict__ wh, _Float16* __restrict__ feat) {
  __shared__ __align__(16) _Float16 Alds[2][64 * 32];    // 2 x 4KB
  __shared__ __align__(16) _Float16 Blds[2][128 * 32];   // 2 x 8KB
  const int mTile = blockIdx.x;       // n*64 + h
  const int nq    = blockIdx.y;       // 0..3 -> co base nq*128
  const int n = mTile >> 6;
  const int h = mTile & 63;
  const int tid  = threadIdx.x;
  const int lane = tid & 31;
  const int wave = tid >> 5;          // 0..7
  const int mslice = wave & 3;        // M sub-slice (16 rows)
  const int nhalf  = wave >> 2;       // which 64-wide half of the 128 N tile

  v8f acc[4] = { {}, {}, {}, {} };

  conv_stageA(Alds[0], x, n, h, 0, tid);
  conv_stageB(Blds[0], wh, nq, 0, tid);
  __syncthreads();

  for (int it = 0; it < 144; ++it) {
    const int cur = it & 1;
    if (it + 1 < 144) {
      conv_stageA(Alds[cur ^ 1], x, n, h, it + 1, tid);
      conv_stageB(Blds[cur ^ 1], wh, nq, it + 1, tid);
    }
    v16h af = load_fragA(Alds[cur], mslice * 16 + (lane & 15), lane);
    #pragma unroll
    for (int t = 0; t < 4; ++t) {
      v16h bf = load_fragB(Blds[cur], nhalf * 64 + t * 16 + (lane & 15), lane);
      acc[t] = __builtin_amdgcn_wmma_f32_16x16x32_f16(false, af, false, bf,
                                                      (short)0, acc[t], false, false);
    }
    __syncthreads();
  }

  // epilogue: bias + ReLU, write feat f16 [pixel][co]
  const int col = lane & 15, mh = lane >> 4;
  #pragma unroll
  for (int t = 0; t < 4; ++t) {
    const int co = nq * 128 + nhalf * 64 + t * 16 + col;
    const float bv = bias[co];
    #pragma unroll
    for (int j = 0; j < 8; ++j) {
      const int pixel = mTile * 64 + mslice * 16 + j + 8 * mh;
      float v = acc[t][j] + bv;
      v = v > 0.f ? v : 0.f;
      feat[(size_t)pixel * 512 + co] = (_Float16)v;
    }
  }
}

// ---------------- fused 1x1 convs (loc 36 + score 18) on WMMA ----------------
// grid 1024, block 128.  N = 64 (padded), K = 512.
// A tile staged via GLOBAL_LOAD_ASYNC_TO_LDS_B128 (ASYNCcnt path).
__global__ __launch_bounds__(128) void heads_wmma_kernel(
    const _Float16* __restrict__ feat, const _Float16* __restrict__ wh2,
    const float* __restrict__ loc_b, const float* __restrict__ score_b,
    float* __restrict__ out) {
  __shared__ __align__(16) _Float16 Alds[64 * 32];
  __shared__ __align__(16) _Float16 Blds[64 * 32];
  const int mTile = blockIdx.x;
  const int tid = threadIdx.x, lane = tid & 31, wave = tid >> 5;

  v8f acc[4] = { {}, {}, {}, {} };

  for (int ck = 0; ck < 16; ++ck) {
    const int cBase = ck * 32;
    { // stage A: contiguous f16 copy, async global -> LDS (per-lane LDS dest)
      const int m = tid >> 1, cseg = tid & 1;
      const _Float16* src = feat + (size_t)(mTile * 64 + m) * 512 + cBase + cseg * 16;
      unsigned lds_off = (unsigned)(size_t)(&Alds[m * 32 + cseg * 16]);
      unsigned long long ga = (unsigned long long)(size_t)src;
      asm volatile("global_load_async_to_lds_b128 %0, %1, off"
                   :: "v"(lds_off), "v"(ga) : "memory");
      asm volatile("global_load_async_to_lds_b128 %0, %1, off offset:16"
                   :: "v"(lds_off), "v"(ga) : "memory");
    }
    { // stage B transposed
      const int c = tid >> 2, jseg = tid & 3;
      const _Float16* src = wh2 + (size_t)(cBase + c) * 64 + jseg * 16;
      v8h p0 = *(const v8h*)src;
      v8h p1 = *(const v8h*)(src + 8);
      #pragma unroll
      for (int i2 = 0; i2 < 8; ++i2) Blds[(jseg * 16 + i2) * 32 + c] = p0[i2];
      #pragma unroll
      for (int i2 = 0; i2 < 8; ++i2) Blds[(jseg * 16 + 8 + i2) * 32 + c] = p1[i2];
    }
    asm volatile("s_wait_asynccnt 0" ::: "memory");
    __syncthreads();
    v16h af = load_fragA(Alds, wave * 16 + (lane & 15), lane);
    #pragma unroll
    for (int t = 0; t < 4; ++t) {
      v16h bf = load_fragB(Blds, t * 16 + (lane & 15), lane);
      acc[t] = __builtin_amdgcn_wmma_f32_16x16x32_f16(false, af, false, bf,
                                                      (short)0, acc[t], false, false);
    }
    __syncthreads();
  }
  // epilogue: scatter into rpn_locs (pixel*36+co) and rpn_scores (pixel*18+co-36)
  const int col = lane & 15, mh = lane >> 4;
  #pragma unroll
  for (int t = 0; t < 4; ++t) {
    const int co = t * 16 + col;
    const bool isLoc = co < 36;
    const bool isSc  = (co >= 36) && (co < 54);
    float bv = 0.f;
    if (isLoc) bv = loc_b[co];
    else if (isSc) bv = score_b[co - 36];
    #pragma unroll
    for (int j = 0; j < 8; ++j) {
      const int pixel = mTile * 64 + wave * 16 + j + 8 * mh;
      const float v = acc[t][j] + bv;
      if (isLoc)     out[LOC_OFF   + (size_t)pixel * 36 + co] = v;
      else if (isSc) out[SCORE_OFF + (size_t)pixel * 18 + (co - 36)] = v;
    }
  }
}

// ---------------- softmax foreground scores ----------------
__global__ void fg_kernel(const float* __restrict__ scores, float* __restrict__ fg) {
  int g = blockIdx.x * blockDim.x + threadIdx.x;
  if (g >= N_ANCH_T) return;
  int pixel = g / 9, a = g - pixel * 9;
  float s0 = scores[(size_t)pixel * 18 + 2 * a];
  float s1 = scores[(size_t)pixel * 18 + 2 * a + 1];
  float m = fmaxf(s0, s1);
  float e0 = __expf(s0 - m), e1 = __expf(s1 - m);
  fg[g] = e1 / (e0 + e1);
}

// ---------------- anchor decode + clip + min-size filter ----------------
__global__ void decode_kernel(const float* __restrict__ out, const float* __restrict__ fg,
                              float* __restrict__ boxes, float* __restrict__ scf,
                              const int* __restrict__ p_h, const int* __restrict__ p_w,
                              const int* __restrict__ p_s) {
  int g = blockIdx.x * blockDim.x + threadIdx.x;
  if (g >= N_ANCH_T) return;
  const float img_h = (float)p_h[0], img_w = (float)p_w[0], scale = (float)p_s[0];
  int n = g / ANCH_PER;
  int k = g - n * ANCH_PER;
  int p = k / 9, a = k - p * 9;
  int pixel = n * 4096 + p;
  int hy = p >> 6, wx = p & 63;
  int i = a / 3, j = a - 3 * i;
  const float ratios[3] = {0.5f, 1.f, 2.f};
  const float scalesv[3] = {8.f, 16.f, 32.f};
  float r = ratios[i], s = scalesv[j];
  float ah = 16.f * s * sqrtf(r);
  float aw = 16.f * s * sqrtf(1.f / r);
  float cy0 = hy * 16.f + 8.f, cx0 = wx * 16.f + 8.f;
  float ay1 = cy0 - 0.5f * ah, ax1 = cx0 - 0.5f * aw;
  float ay2 = cy0 + 0.5f * ah, ax2 = cx0 + 0.5f * aw;
  const float* L = out + LOC_OFF + (size_t)pixel * 36 + a * 4;
  float dy = L[0], dx = L[1], dh = L[2], dw = L[3];
  float sh = ay2 - ay1, sw = ax2 - ax1;
  float cy = ay1 + 0.5f * sh, cx = ax1 + 0.5f * sw;
  float ncy = dy * sh + cy, ncx = dx * sw + cx;
  float nh = __expf(dh) * sh, nw = __expf(dw) * sw;
  float y1 = ncy - 0.5f * nh, x1 = ncx - 0.5f * nw;
  float y2 = ncy + 0.5f * nh, x2 = ncx + 0.5f * nw;
  y1 = fminf(fmaxf(y1, 0.f), img_h); x1 = fminf(fmaxf(x1, 0.f), img_w);
  y2 = fminf(fmaxf(y2, 0.f), img_h); x2 = fminf(fmaxf(x2, 0.f), img_w);
  boxes[(size_t)g * 4 + 0] = y1; boxes[(size_t)g * 4 + 1] = x1;
  boxes[(size_t)g * 4 + 2] = y2; boxes[(size_t)g * 4 + 3] = x2;
  float hs = y2 - y1, wsz = x2 - x1, ms = 16.f * scale;
  float sc = fg[g];
  if (hs < ms || wsz < ms) sc = -__builtin_inff();
  scf[g] = sc;
}

// ---------------- greedy NMS: one workgroup per batch ----------------
__global__ __launch_bounds__(1024) void nms_kernel(const float* __restrict__ boxes,
                                                   float* __restrict__ sc,
                                                   float* __restrict__ rois) {
  const int n = blockIdx.x;
  const float* B = boxes + (size_t)n * ANCH_PER * 4;
  float* S = sc + (size_t)n * ANCH_PER;
  float* R = rois + (size_t)n * POST_NMS * 4;
  __shared__ float sval[1024];
  __shared__ int   sidx[1024];
  __shared__ float sbox[4];
  const int tid = threadIdx.x;
  for (int it = 0; it < POST_NMS; ++it) {
    float bv = -__builtin_inff(); int bi = 0;
    for (int k = tid; k < ANCH_PER; k += 1024) {
      float v = S[k];
      if (v > bv) { bv = v; bi = k; }
    }
    sval[tid] = bv; sidx[tid] = bi;
    __syncthreads();
    for (int s = 512; s > 0; s >>= 1) {
      if (tid < s && sval[tid + s] > sval[tid]) {
        sval[tid] = sval[tid + s]; sidx[tid] = sidx[tid + s];
      }
      __syncthreads();
    }
    const int b = sidx[0];
    if (tid == 0) {
      float y1 = B[b * 4 + 0], x1 = B[b * 4 + 1], y2 = B[b * 4 + 2], x2 = B[b * 4 + 3];
      R[it * 4 + 0] = y1; R[it * 4 + 1] = x1; R[it * 4 + 2] = y2; R[it * 4 + 3] = x2;
      sbox[0] = y1; sbox[1] = x1; sbox[2] = y2; sbox[3] = x2;
    }
    __syncthreads();
    const float by1 = sbox[0], bx1 = sbox[1], by2 = sbox[2], bx2 = sbox[3];
    const float barea = (by2 - by1) * (bx2 - bx1);
    for (int k = tid; k < ANCH_PER; k += 1024) {
      float y1 = B[k * 4 + 0], x1 = B[k * 4 + 1], y2 = B[k * 4 + 2], x2 = B[k * 4 + 3];
      float yy1 = fmaxf(by1, y1), xx1 = fmaxf(bx1, x1);
      float yy2 = fminf(by2, y2), xx2 = fminf(bx2, x2);
      float ih = yy2 - yy1; ih = ih > 0.f ? ih : 0.f;
      float iw = xx2 - xx1; iw = iw > 0.f ? iw : 0.f;
      float inter = ih * iw;
      float area = (y2 - y1) * (x2 - x1);
      float iou = inter / (barea + area - inter + 1e-9f);
      if (iou >= 0.7f || k == b) S[k] = -__builtin_inff();
    }
    __syncthreads();
  }
}

extern "C" void kernel_launch(void* const* d_in, const int* in_sizes, int n_in,
                              void* d_out, int out_size, void* d_ws, size_t ws_size,
                              hipStream_t stream) {
  (void)in_sizes; (void)n_in; (void)out_size; (void)ws_size;
  const float* x        = (const float*)d_in[0];
  const float* conv1_w  = (const float*)d_in[1];
  const float* conv1_b  = (const float*)d_in[2];
  const float* score_w  = (const float*)d_in[3];
  const float* score_b  = (const float*)d_in[4];
  const float* loc_w    = (const float*)d_in[5];
  const float* loc_b    = (const float*)d_in[6];
  const int*   img_h    = (const int*)d_in[7];
  const int*   img_w    = (const int*)d_in[8];
  const int*   scale    = (const int*)d_in[9];
  float* out = (float*)d_out;

  char* ws = (char*)d_ws;
  _Float16* wh    = (_Float16*)(ws + WH_OFF);
  _Float16* wh2   = (_Float16*)(ws + WH2_OFF);
  _Float16* feat  = (_Float16*)(ws + FEAT_OFF);
  float*    fg    = (float*)(ws + FG_OFF);
  float*    boxes = (float*)(ws + BOX_OFF);
  float*    scf   = (float*)(ws + SCF_OFF);

  prep_w_kernel<<<(9 * 512 * 512 + 255) / 256, 256, 0, stream>>>(conv1_w, wh);
  prep_w2_kernel<<<(512 * 64 + 255) / 256, 256, 0, stream>>>(loc_w, score_w, wh2);

  dim3 gconv(1024, 4, 1);
  conv3x3_wmma_kernel<<<gconv, 256, 0, stream>>>(x, conv1_b, wh, feat);

  heads_wmma_kernel<<<1024, 128, 0, stream>>>(feat, wh2, loc_b, score_b, out);

  fg_kernel<<<(N_ANCH_T + 255) / 256, 256, 0, stream>>>(out + SCORE_OFF, fg);
  decode_kernel<<<(N_ANCH_T + 255) / 256, 256, 0, stream>>>(out, fg, boxes, scf,
                                                            img_h, img_w, scale);
  nms_kernel<<<N_BATCH, 1024, 0, stream>>>(boxes, scf, out + ROIS_OFF);
}
